// BP_FNN_Ite_70042326663677
// MI455X (gfx1250) — matrix operand order repaired and checked by reading, not compiled, not run on hardware
//
#include <hip/hip_runtime.h>
#include <hip/hip_bf16.h>

// FNN inference fused as a single bf16 WMMA GEMM:
//   A_aug[b, 0:64]   = x^2      (bf16)
//   A_aug[b, 64:128] = x        (bf16)
//   A_aug[b, 128]    = 1, [129:160) = 0    (K padded to 160 = 5*32)
//   W_all[0:64]    : rule-log coefficients  (-a, 2*a*mu, -sum a*mu^2)
//   W_all[64:192]  : TSK consequents        (0, w3[k,r,f], w3[k,r,64])
//   OUT = A_aug @ W_all^T  -> cols 0:64 = log rule, 64:192 = conq
// Epilogue: stable exp (max-subtract is exact: rule/sum is scale invariant),
// weighted sums over rules, 2-class softmax.

typedef __attribute__((ext_vector_type(16))) __bf16 v16bf;
typedef __attribute__((ext_vector_type(8)))  __bf16 v8bf;
typedef __attribute__((ext_vector_type(8)))  float  v8f;

#define BTILE   64      // rows per workgroup
#define KPAD    160     // padded K (5 * 32)
#define NCOLS   192     // 64 rule cols + 128 conq cols
#define ASTRIDE 168     // LDS A row stride (ushorts), padded vs 160
#define RSTRIDE 193     // LDS result row stride (floats), odd -> no bank conflicts

__device__ __forceinline__ unsigned short f2bf(float f) {
    unsigned int u = __float_as_uint(f);
    unsigned int r = u + 0x7FFFu + ((u >> 16) & 1u);   // round-to-nearest-even
    return (unsigned short)(r >> 16);
}

union Frag16 { v16bf v; v8bf h[2]; };

// ---------------------------------------------------------------- prep kernel
// Builds W_all (bf16, [192][160]) in workspace. One block, 256 threads.
__global__ __launch_bounds__(256) void fnn_prep(const float* __restrict__ mu,
                                                const float* __restrict__ sg,
                                                const float* __restrict__ w3,
                                                unsigned short* __restrict__ W) {
    int n = threadIdx.x;
    if (n < 64) {                       // Gaussian rule-log coefficients
        int r = n;
        unsigned short* wr = W + r * KPAD;
        float c = 0.0f;
        for (int f = 0; f < 64; ++f) {
            float m = mu[r * 64 + f];
            float s = sg[r * 64 + f];
            float a = 1.0f / (2.0f * s * s);      // can reach ~1e7: bf16-safe
            wr[f]      = f2bf(-a);                // * x^2
            wr[64 + f] = f2bf(2.0f * a * m);      // * x
            c += a * m * m;
        }
        wr[128] = f2bf(-c);                       // * 1
        for (int k = 129; k < KPAD; ++k) wr[k] = 0;
    } else if (n < 192) {               // TSK consequent coefficients
        int kr = n - 64;                          // k*64 + r
        const float* src = w3 + kr * 65;          // para_w3 is (2,64,65)
        unsigned short* wr = W + n * KPAD;
        for (int f = 0; f < 64; ++f) {
            wr[f]      = 0;                       // no x^2 term
            wr[64 + f] = f2bf(src[f]);            // * x
        }
        wr[128] = f2bf(src[64]);                  // bias (* 1)
        for (int k = 129; k < KPAD; ++k) wr[k] = 0;
    }
}

// ---------------------------------------------------------------- main kernel
__global__ __launch_bounds__(256) void fnn_main(const float* __restrict__ data,
                                                const unsigned short* __restrict__ W,
                                                float* __restrict__ out) {
    // LDS aliased across phases: bf16 A tile (64x168 = 21504 B), then
    // f32 result tile (64x193 = 49408 B). Barriers separate the uses.
    __shared__ __align__(16) unsigned char smem[BTILE * RSTRIDE * 4];
    unsigned short* A   = (unsigned short*)smem;   // [64][ASTRIDE]
    float*          RES = (float*)smem;            // [64][RSTRIDE]

    const int tid  = threadIdx.x;
    const int row0 = blockIdx.x * BTILE;

    // ---- Phase 1: build augmented bf16 A tile in LDS (4 threads per row)
    {
        int row = tid >> 2, seg = tid & 3;
        const float* src = data + (size_t)(row0 + row) * 64 + seg * 16;
        float x[16];
#pragma unroll
        for (int i = 0; i < 4; ++i) {
            float4 v = reinterpret_cast<const float4*>(src)[i];
            x[i * 4 + 0] = v.x; x[i * 4 + 1] = v.y;
            x[i * 4 + 2] = v.z; x[i * 4 + 3] = v.w;
        }
        unsigned short* arow = A + row * ASTRIDE;
#pragma unroll
        for (int i = 0; i < 16; ++i) {
            int f = seg * 16 + i;
            arow[f]      = f2bf(x[i] * x[i]);
            arow[64 + f] = f2bf(x[i]);
        }
        if (seg == 3) {
            arow[128] = 0x3F80;                    // 1.0 in bf16
            for (int k = 129; k < KPAD; ++k) arow[k] = 0;
        }
    }
    __syncthreads();

    // ---- Phase 2: WMMA GEMM. 8 waves: wave = (row-tile rt 0..3, col-half ch 0..1)
    const int lane    = tid & 31;
    const int wv      = tid >> 5;
    const int rt      = wv & 3;            // rows rt*16 .. rt*16+15
    const int ch      = wv >> 2;           // cols ch*96 .. ch*96+95 (6 N-tiles)
    const int laneRow = lane & 15;
    const int koffA   = (lane >> 4) * 8;   // A: hi lanes hold K+8 / K+24
    const int koffB   = (lane >> 4) * 8;   // B: hi lanes hold K+8 / K+24

    v8f acc[6] = {};
    const unsigned short* arow = A + (rt * 16 + laneRow) * ASTRIDE;

#pragma unroll
    for (int kk = 0; kk < KPAD / 32; ++kk) {
        const int k0 = kk * 32;
        // A fragment per ISA 16-bit 16x32 layout (two 16B LDS loads)
        Frag16 fa;
        fa.h[0] = *reinterpret_cast<const v8bf*>(arow + k0 + koffA);
        fa.h[1] = *reinterpret_cast<const v8bf*>(arow + k0 + 16 + koffA);
#pragma unroll
        for (int nt = 0; nt < 6; ++nt) {
            const int ncol = ch * 96 + nt * 16 + laneRow;
            const unsigned short* wp = W + (size_t)ncol * KPAD + k0 + koffB;
            Frag16 fb;                     // B: lane = column, same K order as A
            fb.h[0] = *reinterpret_cast<const v8bf*>(wp);
            fb.h[1] = *reinterpret_cast<const v8bf*>(wp + 16);
            acc[nt] = __builtin_amdgcn_wmma_f32_16x16x32_bf16(
                false, fa.v, false, fb.v, (short)0, acc[nt], false, false);
        }
    }
    __syncthreads();   // all LDS A reads done before aliasing as RES

    // ---- Phase 3: spill accumulators to LDS as f32 [64][193]
    {
        const int rbase = rt * 16 + (lane >> 4) * 8;   // C layout: VGPR j -> M=j / j+8
#pragma unroll
        for (int nt = 0; nt < 6; ++nt) {
            const int col = ch * 96 + nt * 16 + laneRow;
#pragma unroll
            for (int j = 0; j < 8; ++j)
                RES[(rbase + j) * RSTRIDE + col] = acc[nt][j];
        }
    }
    __syncthreads();

    // ---- Phase 4: epilogue, one thread per output row
    if (tid < BTILE) {
        const float* r = RES + tid * RSTRIDE;
        float m = r[0];
        for (int i = 1; i < 64; ++i) m = fmaxf(m, r[i]);
        float S = 0.0f, d0 = 0.0f, d1 = 0.0f;
        for (int i = 0; i < 64; ++i) {
            float e = __expf(r[i] - m);        // scale cancels in rule/S
            S  += e;
            d0 += e * r[64 + i];               // conq k=0
            d1 += e * r[128 + i];              // conq k=1
        }
        float l0 = d0 / S, l1 = d1 / S;
        float mm = fmaxf(l0, l1);
        float e0 = __expf(l0 - mm), e1 = __expf(l1 - mm);
        float inv = 1.0f / (e0 + e1);
        out[(size_t)(row0 + tid) * 2 + 0] = e0 * inv;
        out[(size_t)(row0 + tid) * 2 + 1] = e1 * inv;
    }
}

// ---------------------------------------------------------------- launch
extern "C" void kernel_launch(void* const* d_in, const int* in_sizes, int n_in,
                              void* d_out, int out_size, void* d_ws, size_t ws_size,
                              hipStream_t stream) {
    const float* data = (const float*)d_in[0];   // [B,64] f32
    const float* mu   = (const float*)d_in[1];   // [64,64] f32
    const float* sg   = (const float*)d_in[2];   // [64,64] f32
    const float* w3   = (const float*)d_in[3];   // [2,64,65] f32
    float* out = (float*)d_out;
    unsigned short* W = (unsigned short*)d_ws;   // needs 192*160*2 = 61440 B

    const int B = in_sizes[0] / 64;              // 16384
    fnn_prep<<<1, 256, 0, stream>>>(mu, sg, w3, W);
    fnn_main<<<B / BTILE, 256, 0, stream>>>(data, W, out);
    (void)n_in; (void)out_size; (void)ws_size;
}
